// MultiHeadAttention_6021544149498
// MI455X (gfx1250) — matrix operand order repaired and checked by reading, not compiled, not run on hardware
//
#include <hip/hip_runtime.h>
#include <hip/hip_bf16.h>

// ---------------------------------------------------------------------------
// MI455X (gfx1250, wave32) multi-head attention, all GEMMs via
// v_wmma_f32_16x16x32_f16 (fp32 accumulate).
//  - explicit ping-pong double buffering (no register-rotation moves)
//  - shuffle-free online softmax (per-lane m/l, single end-merge)
//  - NT stores for the 537MB weights stream, global_prefetch for K tiles
//
// D_MODEL=1024, H=16, depth=64, B=2, S=2048.
// d_out = [ out (2*2048*1024 f32) | weights (2*16*2048*2048 f32) ]
// ---------------------------------------------------------------------------

typedef __attribute__((ext_vector_type(16))) _Float16 v16h;
typedef __attribute__((ext_vector_type(8)))  _Float16 v8h;
typedef __attribute__((ext_vector_type(8)))  float    v8f;
typedef __attribute__((ext_vector_type(4)))  float    v4f;

#define WMMA_F16(A, B, C) \
  __builtin_amdgcn_wmma_f32_16x16x32_f16(false, (A), false, (B), (short)0, (C), false, false)

// ---- WMMA tile loaders (wave32 layouts per CDNA5 ISA 7.12.2) ---------------
// A 16x32 f16: lane<16 -> row=lane,   K {0..7, 16..23}
//              lane>=16 -> row=lane-16, K {8..15, 24..31}
__device__ __forceinline__ v16h load_a_f16(const _Float16* base, int ld, int lane) {
  const int r  = lane & 15;
  const int kb = (lane >> 4) * 8;
  const _Float16* p = base + (size_t)r * ld + kb;
  v8h lo = *(const v8h*)(p);
  v8h hi = *(const v8h*)(p + 16);
  return __builtin_shufflevector(lo, hi, 0,1,2,3,4,5,6,7,8,9,10,11,12,13,14,15);
}

// Same A tile, but source is fp32 row-major (converted on the fly).
__device__ __forceinline__ v16h load_a_f32(const float* base, int ld, int lane) {
  const int r  = lane & 15;
  const int kb = (lane >> 4) * 8;
  const float* p = base + (size_t)r * ld + kb;
  v4f x0 = *(const v4f*)(p);
  v4f x1 = *(const v4f*)(p + 4);
  v4f x2 = *(const v4f*)(p + 16);
  v4f x3 = *(const v4f*)(p + 20);
  v16h a;
#pragma unroll
  for (int i = 0; i < 4; ++i) {
    a[i]      = (_Float16)x0[i];
    a[4 + i]  = (_Float16)x1[i];
    a[8 + i]  = (_Float16)x2[i];
    a[12 + i] = (_Float16)x3[i];
  }
  return a;
}

// B 32x16 f16 from [N][K] storage: lane n<16 -> col n, K {0..7, 16..23};
// lane>=16 -> col n-16, K {8..15, 24..31}.  Two 16B loads per lane.
__device__ __forceinline__ v16h load_b_f16(const _Float16* base, int ld, int lane) {
  const int n  = lane & 15;
  const int kb = (lane >> 4) * 8;
  const _Float16* p = base + (size_t)n * ld + kb;
  v8h lo = *(const v8h*)(p);
  v8h hi = *(const v8h*)(p + 16);
  return __builtin_shufflevector(lo, hi, 0,1,2,3,4,5,6,7,8,9,10,11,12,13,14,15);
}

// Per-lane online softmax update over one score tile (no cross-lane ops).
__device__ __forceinline__ void online_update(const v8f& c, float scale,
                                              float* m, float* l) {
#pragma unroll
  for (int j = 0; j < 8; ++j) {
    const float s  = c[j] * scale;
    const float mn = fmaxf(m[j], s);
    l[j] = l[j] * __expf(m[j] - mn) + __expf(s - mn);
    m[j] = mn;
  }
}

// ---------------------------------------------------------------------------
// Kernel 0: weight prep — W [K=1024][N=1024] fp32 -> Wt [N][K] f16
// ---------------------------------------------------------------------------
__global__ __launch_bounds__(256) void prep_w(const float* __restrict__ W,
                                              _Float16* __restrict__ Wt) {
  int idx = blockIdx.x * 256 + threadIdx.x;   // 0 .. 1024*1024-1
  int n = idx & 1023;
  int k = idx >> 10;
  Wt[(size_t)n * 1024 + k] = (_Float16)W[(size_t)k * 1024 + n];
}

// ---------------------------------------------------------------------------
// Kernel 1: projection  Y = X @ W + b,  head-split f16 output.
// grid (256 m-tiles, 16 heads), block 32 (1 wave). Ping-pong pipelined.
// transpose_out==0 : Out[B,H,S,64]  (Q, K);  ==1 : Out[B,H,64,S]  (V^T)
// ---------------------------------------------------------------------------
__global__ __launch_bounds__(32) void qkv_proj(const float* __restrict__ X,
                                               const _Float16* __restrict__ Wt,
                                               const float* __restrict__ bias,
                                               _Float16* __restrict__ Out,
                                               int transpose_out) {
  const int lane = threadIdx.x;
  const int mt   = blockIdx.x;   // 16-row tile of the 4096-row input
  const int hd   = blockIdx.y;   // head -> 64 output columns

  v8f acc[4] = {};
  const float* Xrow = X + (size_t)mt * 16 * 1024;
  const _Float16* Wb = Wt + (size_t)(hd * 64) * 1024;

  // ping-pong buffers: A set holds even k-steps, B set odd k-steps
  v16h aA = load_a_f32(Xrow, 1024, lane);
  v16h bA[4], aB, bB[4];
#pragma unroll
  for (int t = 0; t < 4; ++t)
    bA[t] = load_b_f16(Wb + (size_t)t * 16 * 1024, 1024, lane);

#pragma unroll 1
  for (int ks = 0; ks < 30; ks += 2) {         // K = 1024 in steps of 32
    aB = load_a_f32(Xrow + (ks + 1) * 32, 1024, lane);
#pragma unroll
    for (int t = 0; t < 4; ++t)
      bB[t] = load_b_f16(Wb + (size_t)t * 16 * 1024 + (ks + 1) * 32, 1024, lane);
#pragma unroll
    for (int t = 0; t < 4; ++t)
      acc[t] = WMMA_F16(aA, bA[t], acc[t]);

    aA = load_a_f32(Xrow + (ks + 2) * 32, 1024, lane);
#pragma unroll
    for (int t = 0; t < 4; ++t)
      bA[t] = load_b_f16(Wb + (size_t)t * 16 * 1024 + (ks + 2) * 32, 1024, lane);
#pragma unroll
    for (int t = 0; t < 4; ++t)
      acc[t] = WMMA_F16(aB, bB[t], acc[t]);
  }
  // tail: k-steps 30 (in A set) and 31
  aB = load_a_f32(Xrow + 31 * 32, 1024, lane);
#pragma unroll
  for (int t = 0; t < 4; ++t)
    bB[t] = load_b_f16(Wb + (size_t)t * 16 * 1024 + 31 * 32, 1024, lane);
#pragma unroll
  for (int t = 0; t < 4; ++t) acc[t] = WMMA_F16(aA, bA[t], acc[t]);
#pragma unroll
  for (int t = 0; t < 4; ++t) acc[t] = WMMA_F16(aB, bB[t], acc[t]);

  const int colq  = lane & 15;
  const int rbase = (lane >> 4) * 8;
#pragma unroll
  for (int t = 0; t < 4; ++t) {
    const float bval = bias[hd * 64 + t * 16 + colq];
#pragma unroll
    for (int j = 0; j < 8; ++j) {
      const int grow = mt * 16 + rbase + j;    // row in [0, 4096)
      const int b_   = grow >> 11;             // batch
      const int s_   = grow & 2047;            // sequence pos
      const int d_   = t * 16 + colq;          // 0..63
      const float vv = acc[t][j] + bval;
      if (!transpose_out)
        Out[(((size_t)(b_ * 16 + hd) * 2048) + s_) * 64 + d_] = (_Float16)vv;
      else
        Out[(((size_t)(b_ * 16 + hd) * 64) + d_) * 2048 + s_] = (_Float16)vv;
    }
  }
}

// ---------------------------------------------------------------------------
// Kernel 2: fused attention. grid (128 q-tiles, 16 heads, 2 batch), 1 wave.
// Pass 1: shuffle-free online max/sum per lane over 128 key tiles, then one
//         4-step (m,l) merge across the 16-lane half.
// Pass 2: recompute scores, normalize, NT-write weights once, LDS-transpose
//         P into A-layout, accumulate P@V with WMMA (V pre-transposed).
// ---------------------------------------------------------------------------
__global__ __launch_bounds__(32) void attention_kernel(
    const _Float16* __restrict__ Qh, const _Float16* __restrict__ Kh,
    const _Float16* __restrict__ Vth, float* __restrict__ weights,
    _Float16* __restrict__ concat) {
  __shared__ float pl[16][36];                 // padded 16x32 P staging tile

  const int lane = threadIdx.x;
  const int qt = blockIdx.x, h = blockIdx.y, b = blockIdx.z;
  const size_t bh = (size_t)b * 16 + h;

  const _Float16* Q  = Qh + bh * 2048 * 64 + (size_t)qt * 16 * 64;
  const _Float16* K  = Kh + bh * 2048 * 64;
  const _Float16* Vt = Vth + bh * 64 * 2048;

  // Q tile (16 x 64) kept resident: two K=32 A-tiles
  const v16h a0 = load_a_f16(Q, 64, lane);
  const v16h a1 = load_a_f16(Q + 32, 64, lane);

  const float scale = 0.125f;                  // 1/sqrt(64)
  float m[8], l[8];
#pragma unroll
  for (int j = 0; j < 8; ++j) { m[j] = -3.0e38f; l[j] = 0.f; }

  // ---------------- pass 1: per-lane softmax statistics ----------------
  v16h b0A = load_b_f16(K, 64, lane);
  v16h b1A = load_b_f16(K + 32, 64, lane);
  v16h b0B, b1B;
#pragma unroll 1
  for (int kt = 0; kt < 126; kt += 2) {
    const _Float16* K1 = K + (size_t)(kt + 1) * 1024;
    b0B = load_b_f16(K1, 64, lane);
    b1B = load_b_f16(K1 + 32, 64, lane);
    __builtin_prefetch(K + (size_t)min(kt + 8, 127) * 1024 + (lane << 5), 0, 1);
    v8f c = {};
    c = WMMA_F16(a0, b0A, c);
    c = WMMA_F16(a1, b1A, c);
    online_update(c, scale, m, l);

    const _Float16* K2 = K + (size_t)(kt + 2) * 1024;
    b0A = load_b_f16(K2, 64, lane);
    b1A = load_b_f16(K2 + 32, 64, lane);
    __builtin_prefetch(K + (size_t)min(kt + 9, 127) * 1024 + (lane << 5), 0, 1);
    c = v8f{};
    c = WMMA_F16(a0, b0B, c);
    c = WMMA_F16(a1, b1B, c);
    online_update(c, scale, m, l);
  }
  {                                            // tail: kt = 126, 127
    const _Float16* K1 = K + (size_t)127 * 1024;
    b0B = load_b_f16(K1, 64, lane);
    b1B = load_b_f16(K1 + 32, 64, lane);
    v8f c = {};
    c = WMMA_F16(a0, b0A, c);
    c = WMMA_F16(a1, b1A, c);
    online_update(c, scale, m, l);
    c = v8f{};
    c = WMMA_F16(a0, b0B, c);
    c = WMMA_F16(a1, b1B, c);
    online_update(c, scale, m, l);
  }

  // merge (m,l) across the 16-lane half: all lanes end with full row stats
  float invl[8];
#pragma unroll
  for (int j = 0; j < 8; ++j) {
#pragma unroll
    for (int off = 1; off < 16; off <<= 1) {
      const float mo = __shfl_xor(m[j], off);
      const float lo = __shfl_xor(l[j], off);
      const float M  = fmaxf(m[j], mo);
      l[j] = l[j] * __expf(m[j] - M) + lo * __expf(mo - M);
      m[j] = M;
    }
    invl[j] = 1.0f / l[j];
  }

  // ---------------- pass 2: weights out + P@V ----------------
  v8f av[4] = {};
  float* wout = weights + (bh * 2048 + (size_t)qt * 16) * 2048;
  const int colq = lane & 15;
  const int rb   = (lane >> 4) * 8;

  for (int kp = 0; kp < 64; ++kp) {            // 32 keys per step
    // issue V tiles early: they overlap the score WMMAs / exp / stores
    v16h bv[4];
#pragma unroll
    for (int t = 0; t < 4; ++t)
      bv[t] = load_b_f16(Vt + (size_t)(t * 16) * 2048 + kp * 32, 2048, lane);

#pragma unroll
    for (int half = 0; half < 2; ++half) {
      const int kt = kp * 2 + half;
      const _Float16* Kt = K + (size_t)kt * 1024;
      v8f c = {};
      c = WMMA_F16(a0, load_b_f16(Kt, 64, lane), c);
      c = WMMA_F16(a1, load_b_f16(Kt + 32, 64, lane), c);
#pragma unroll
      for (int j = 0; j < 8; ++j) {
        const int row = rb + j;
        const float p = __expf(c[j] * scale - m[j]) * invl[j];
        // streaming store: written once, never re-read -> keep L2 for K/V
        __builtin_nontemporal_store(p, &wout[(size_t)row * 2048 + kt * 16 + colq]);
        pl[row][half * 16 + colq] = p;         // stage for lane transpose
      }
    }
    __syncthreads();
    // Re-read P in A-layout (lane=row), convert to f16.
    const int r  = lane & 15;
    const int kb = (lane >> 4) * 8;
    v16h ap;
#pragma unroll
    for (int i = 0; i < 8; ++i) {
      ap[i]     = (_Float16)pl[r][kb + i];
      ap[8 + i] = (_Float16)pl[r][kb + 16 + i];
    }
    __syncthreads();
#pragma unroll
    for (int t = 0; t < 4; ++t)                // 64 dims = 4 N-tiles
      av[t] = WMMA_F16(ap, bv[t], av[t]);
  }

  // concat [B, S, H*64] in f16 for the output projection
  _Float16* cc = concat + ((size_t)b * 2048 + qt * 16) * 1024 + h * 64;
#pragma unroll
  for (int t = 0; t < 4; ++t)
#pragma unroll
    for (int j = 0; j < 8; ++j)
      cc[(size_t)(rb + j) * 1024 + t * 16 + colq] = (_Float16)av[t][j];
}

// ---------------------------------------------------------------------------
// Kernel 3: out = concat @ wo + bo (fp32 out). grid (256, 16), 1 wave.
// ---------------------------------------------------------------------------
__global__ __launch_bounds__(32) void out_proj(const _Float16* __restrict__ Xh,
                                               const _Float16* __restrict__ Wt,
                                               const float* __restrict__ bias,
                                               float* __restrict__ Out) {
  const int lane = threadIdx.x;
  const int mt = blockIdx.x, ns = blockIdx.y;  // ns: strip of 64 columns
  v8f acc[4] = {};
  const _Float16* Xrow = Xh + (size_t)mt * 16 * 1024;
  const _Float16* Wb   = Wt + (size_t)(ns * 64) * 1024;

  v16h aA = load_a_f16(Xrow, 1024, lane);
  v16h bA[4], aB, bB[4];
#pragma unroll
  for (int t = 0; t < 4; ++t)
    bA[t] = load_b_f16(Wb + (size_t)t * 16 * 1024, 1024, lane);

#pragma unroll 1
  for (int ks = 0; ks < 30; ks += 2) {
    aB = load_a_f16(Xrow + (ks + 1) * 32, 1024, lane);
#pragma unroll
    for (int t = 0; t < 4; ++t)
      bB[t] = load_b_f16(Wb + (size_t)t * 16 * 1024 + (ks + 1) * 32, 1024, lane);
#pragma unroll
    for (int t = 0; t < 4; ++t)
      acc[t] = WMMA_F16(aA, bA[t], acc[t]);

    aA = load_a_f16(Xrow + (ks + 2) * 32, 1024, lane);
#pragma unroll
    for (int t = 0; t < 4; ++t)
      bA[t] = load_b_f16(Wb + (size_t)t * 16 * 1024 + (ks + 2) * 32, 1024, lane);
#pragma unroll
    for (int t = 0; t < 4; ++t)
      acc[t] = WMMA_F16(aB, bB[t], acc[t]);
  }
  aB = load_a_f16(Xrow + 31 * 32, 1024, lane);
#pragma unroll
  for (int t = 0; t < 4; ++t)
    bB[t] = load_b_f16(Wb + (size_t)t * 16 * 1024 + 31 * 32, 1024, lane);
#pragma unroll
  for (int t = 0; t < 4; ++t) acc[t] = WMMA_F16(aA, bA[t], acc[t]);
#pragma unroll
  for (int t = 0; t < 4; ++t) acc[t] = WMMA_F16(aB, bB[t], acc[t]);

  const int colq = lane & 15;
  const int rb   = (lane >> 4) * 8;
#pragma unroll
  for (int t = 0; t < 4; ++t) {
    const float bval = bias[ns * 64 + t * 16 + colq];
#pragma unroll
    for (int j = 0; j < 8; ++j)
      Out[(size_t)(mt * 16 + rb + j) * 1024 + ns * 64 + t * 16 + colq] =
          acc[t][j] + bval;
  }
}

// ---------------------------------------------------------------------------
extern "C" void kernel_launch(void* const* d_in, const int* in_sizes, int n_in,
                              void* d_out, int out_size, void* d_ws, size_t ws_size,
                              hipStream_t stream) {
  const float* queries = (const float*)d_in[0];
  const float* keys    = (const float*)d_in[1];
  const float* values  = (const float*)d_in[2];
  const float* wq = (const float*)d_in[3];
  const float* bq = (const float*)d_in[4];
  const float* wk = (const float*)d_in[5];
  const float* bk = (const float*)d_in[6];
  const float* wv = (const float*)d_in[7];
  const float* bv = (const float*)d_in[8];
  const float* wo = (const float*)d_in[9];
  const float* bo = (const float*)d_in[10];

  // Workspace layout (bytes):
  //   0   Wt_q | 2MB Wt_k | 4MB Wt_v | 6MB Wt_o          (f16 [N][K], 2MB each)
  //   8MB  Qh  [B,H,S,64] f16   | 16MB Kh [B,H,S,64] f16
  //   24MB Vth [B,H,64,S] f16   | 32MB concat [B,S,1024] f16  -> total 40MB
  const size_t MB = 1u << 20;
  if (ws_size < 40 * MB) return;
  char* ws = (char*)d_ws;
  _Float16* Wtq = (_Float16*)(ws + 0 * MB);
  _Float16* Wtk = (_Float16*)(ws + 2 * MB);
  _Float16* Wtv = (_Float16*)(ws + 4 * MB);
  _Float16* Wto = (_Float16*)(ws + 6 * MB);
  _Float16* Qh  = (_Float16*)(ws + 8 * MB);
  _Float16* Kh  = (_Float16*)(ws + 16 * MB);
  _Float16* Vth = (_Float16*)(ws + 24 * MB);
  _Float16* Cc  = (_Float16*)(ws + 32 * MB);

  float* out      = (float*)d_out;
  float* weightsO = out + (size_t)2 * 2048 * 1024;   // tuple: (out, weights)

  // 0) weight transpose+convert
  prep_w<<<dim3(4096), dim3(256), 0, stream>>>(wq, Wtq);
  prep_w<<<dim3(4096), dim3(256), 0, stream>>>(wk, Wtk);
  prep_w<<<dim3(4096), dim3(256), 0, stream>>>(wv, Wtv);
  prep_w<<<dim3(4096), dim3(256), 0, stream>>>(wo, Wto);

  // 1) QKV projections (WMMA)
  qkv_proj<<<dim3(256, 16), dim3(32), 0, stream>>>(queries, Wtq, bq, Qh, 0);
  qkv_proj<<<dim3(256, 16), dim3(32), 0, stream>>>(keys,    Wtk, bk, Kh, 0);
  qkv_proj<<<dim3(256, 16), dim3(32), 0, stream>>>(values,  Wtv, bv, Vth, 1);

  // 2) fused scores/softmax/weights/P@V (WMMA)
  attention_kernel<<<dim3(128, 16, 2), dim3(32), 0, stream>>>(Qh, Kh, Vth,
                                                              weightsO, Cc);

  // 3) output projection (WMMA)
  out_proj<<<dim3(256, 16), dim3(32), 0, stream>>>(Cc, Wto, bo, out);
}
